// TokenFlowTokenizer_62603443307091
// MI455X (gfx1250) — matrix operand order repaired and checked by reading, not compiled, not run on hardware
//
#include <hip/hip_runtime.h>

// ===========================================================================
// VQ-VAE tokenizer for MI455X (gfx1250, wave32, WMMA bf16/f32-accum).
// NHWC bf16 activations + channel-minor K ordering => A fragments are two
// aligned 16B vector loads (branch-free zero padding). Weights/codebook
// pre-packed into exact B-fragment layout (coalesced 32B loads).
// K-loops are software-pipelined; the VQ loop uses a 2-deep B prefetch so
// waits before WMMA are partial (s_wait_loadcnt 2), not full stalls.
// ===========================================================================

typedef __bf16 bf16_t;
typedef __attribute__((ext_vector_type(16))) __bf16        v16bf;
typedef __attribute__((ext_vector_type(8)))  float         v8f;
typedef __attribute__((ext_vector_type(4)))  unsigned int  v4u;
typedef __attribute__((ext_vector_type(8)))  unsigned int  v8u;

__device__ __forceinline__ bf16_t f2bf(float f) {
  unsigned u = __float_as_uint(f);
  u = u + 0x7FFFu + ((u >> 16) & 1u);          // round-to-nearest-even
  unsigned short s = (unsigned short)(u >> 16);
  return __builtin_bit_cast(bf16_t, s);
}

// Branch-free "load 8 bf16 or zeros": select offset, always load, mask result.
__device__ __forceinline__ v4u load16_masked(const bf16_t* base, int off, bool ok) {
  unsigned msk = ok ? 0xFFFFFFFFu : 0u;
  int o = ok ? off : 0;
  v4u r = *(const v4u*)(base + o);
  r &= msk;
  return r;
}
__device__ __forceinline__ v16bf combine(v4u r0, v4u r1) {
  v8u c = __builtin_shufflevector(r0, r1, 0, 1, 2, 3, 4, 5, 6, 7);
  return __builtin_bit_cast(v16bf, c);
}

// ---------------------------------------------------------------------------
// x [16,3,256,256] f32 NCHW -> xb [16,256,256,8] bf16 NHWC (channels padded)
// ---------------------------------------------------------------------------
__global__ void cvt_x_nhwc(const float* __restrict__ x, bf16_t* __restrict__ xb, int n) {
  int i = blockIdx.x * blockDim.x + threadIdx.x;
  int stride = gridDim.x * blockDim.x;
  for (; i < n; i += stride) {
    int c = i & 7;
    int p = i >> 3;                   // b*65536 + h*256 + w
    int b = p >> 16, hw = p & 65535;
    float v = (c < 3) ? x[((size_t)b * 3 + c) * 65536 + hw] : 0.f;
    xb[i] = f2bf(v);
  }
}

// ---------------------------------------------------------------------------
// Pack stride-2 conv weights (OIHW f32) into B-fragment bf16 layout.
// K ordering: k = (kh*4+kw)*CiPad + ci.  grid.x = (K/32)*(Co/16), block=32
// ---------------------------------------------------------------------------
__global__ __launch_bounds__(32)
void pack_conv_w(const float* __restrict__ w, bf16_t* __restrict__ pack,
                 int ciShift, int CiReal, int Co, int K) {
  const int lane = threadIdx.x;
  const int CiPad = 1 << ciShift;
  const int nCoT = Co >> 4;
  const int kt = blockIdx.x / nCoT;
  const int ct = blockIdx.x % nCoT;
  const int n = lane & 15;
  const int base16 = (lane >> 4) * 16;
  const int co = ct * 16 + n;
  v16bf frag;
#pragma unroll
  for (int e = 0; e < 16; ++e) {
    int k = kt * 32 + (base16 >> 1) + (e & 7) + ((e >> 3) << 4);
    int tap = k >> ciShift, ci = k & (CiPad - 1);
    float v = 0.f;
    if (k < K && ci < CiReal) {
      int kh = tap >> 2, kw = tap & 3;
      v = w[(((size_t)co * CiReal + ci) * 4 + kh) * 4 + kw];
    }
    frag[e] = f2bf(v);
  }
  *(v16bf*)(pack + ((size_t)blockIdx.x * 32 + lane) * 16) = frag;
}

// ---------------------------------------------------------------------------
// Pack deconv weights per output parity class p. K ordering:
// k = (khs*2+kws)*Ci + ci, kh = ph+2*khs, kw = pw+2*kws.
// grid.x = 4*(K/32)*ceil(Co/16)
// ---------------------------------------------------------------------------
__global__ __launch_bounds__(32)
void pack_deconv_w(const float* __restrict__ w, bf16_t* __restrict__ pack,
                   int ciShift, int Co, int K) {
  const int lane = threadIdx.x;
  const int Ci = 1 << ciShift;
  const int nCoT = (Co + 15) >> 4;
  const int kT = K >> 5;
  int bx = blockIdx.x;
  const int p = bx / (kT * nCoT);
  int rem = bx % (kT * nCoT);
  const int kt = rem / nCoT;
  const int ct = rem % nCoT;
  const int ph = p >> 1, pw = p & 1;
  const int n = lane & 15;
  const int base16 = (lane >> 4) * 16;
  const int co = ct * 16 + n;
  v16bf frag;
#pragma unroll
  for (int e = 0; e < 16; ++e) {
    int k = kt * 32 + (base16 >> 1) + (e & 7) + ((e >> 3) << 4);
    int tap = k >> ciShift, ci = k & (Ci - 1);
    float v = 0.f;
    if (co < Co) {
      int kh = ph + 2 * (tap >> 1), kw = pw + 2 * (tap & 1);
      v = w[(((size_t)co * Ci + ci) * 4 + kh) * 4 + kw];
    }
    frag[e] = f2bf(v);
  }
  *(v16bf*)(pack + ((size_t)bx * 32 + lane) * 16) = frag;
}

// ---------------------------------------------------------------------------
// Pack codebook [8192,256] f32 into B-fragment layout. grid.x = 8*512
// ---------------------------------------------------------------------------
__global__ __launch_bounds__(32)
void pack_cb(const float* __restrict__ cb, bf16_t* __restrict__ pack) {
  const int lane = threadIdx.x;
  const int kt = blockIdx.x >> 9;
  const int ct = blockIdx.x & 511;
  const int n = lane & 15;
  const int base16 = (lane >> 4) * 16;
  const int code = ct * 16 + n;
  v16bf frag;
#pragma unroll
  for (int e = 0; e < 16; ++e) {
    int k = kt * 32 + (base16 >> 1) + (e & 7) + ((e >> 3) << 4);
    frag[e] = f2bf(cb[(size_t)code * 256 + k]);
  }
  *(v16bf*)(pack + ((size_t)blockIdx.x * 32 + lane) * 16) = frag;
}

// ---------------------------------------------------------------------------
// Implicit-GEMM stride-2 conv (k4 s2 p1), NHWC bf16, WMMA, pipelined k-loop.
// grid.x = totalMtiles/4, grid.y = (Co/16)/NT, block = 128 (4 waves)
// ---------------------------------------------------------------------------
template <int NT>
__global__ __launch_bounds__(128)
void conv2d_wmma(const bf16_t* __restrict__ in, const bf16_t* __restrict__ wpack,
                 const float* __restrict__ bias, bf16_t* __restrict__ out,
                 int ciShift, int Hi, int Wi, int Ho, int woShift,
                 int tpiShift, int nCoT, int coShift, int kTiles, int relu) {
  const int lane = threadIdx.x & 31;
  const int mt = blockIdx.x * 4 + (threadIdx.x >> 5);
  const int Wo = 1 << woShift;
  const int Ci = 1 << ciShift;
  const int b = mt >> tpiShift;
  const int tloc = mt & ((1 << tpiShift) - 1);
  const int ctBase = blockIdx.y * NT;
  const int m = lane & 15;
  const int pos = tloc * 16 + m;
  const int oh = pos >> woShift, ow = pos & (Wo - 1);
  const bf16_t* inB = in + (((size_t)b * Hi) << ciShift) * Wi;
  const int base8 = (lane >> 4) * 8;

  auto loadA = [&](int kt) -> v16bf {
    const int s0 = kt * 32 + base8;
    const int s1 = s0 + 16;
    int tap0 = s0 >> ciShift, ci0 = s0 & (Ci - 1);
    int ih0 = 2 * oh - 1 + (tap0 >> 2), iw0 = 2 * ow - 1 + (tap0 & 3);
    bool ok0 = (unsigned)ih0 < (unsigned)Hi && (unsigned)iw0 < (unsigned)Wi;
    v4u r0 = load16_masked(inB, ((ih0 * Wi + iw0) << ciShift) + ci0, ok0);
    int tap1 = s1 >> ciShift, ci1 = s1 & (Ci - 1);
    int ih1 = 2 * oh - 1 + (tap1 >> 2), iw1 = 2 * ow - 1 + (tap1 & 3);
    bool ok1 = (unsigned)ih1 < (unsigned)Hi && (unsigned)iw1 < (unsigned)Wi;
    v4u r1 = load16_masked(inB, ((ih1 * Wi + iw1) << ciShift) + ci1, ok1);
    return combine(r0, r1);
  };
  auto loadB = [&](int kt, int j) -> v16bf {
    return *(const v16bf*)(wpack + ((size_t)(kt * nCoT + ctBase + j) * 32 + lane) * 16);
  };

  v8f acc[NT];
#pragma unroll
  for (int j = 0; j < NT; ++j)
#pragma unroll
    for (int i = 0; i < 8; ++i) acc[j][i] = 0.f;

  // software pipeline: prefetch depth 1
  v16bf a = loadA(0);
  v16bf bfr[NT];
#pragma unroll
  for (int j = 0; j < NT; ++j) bfr[j] = loadB(0, j);

  for (int kt = 0; kt < kTiles - 1; ++kt) {
    v16bf an = loadA(kt + 1);
    v16bf bn[NT];
#pragma unroll
    for (int j = 0; j < NT; ++j) bn[j] = loadB(kt + 1, j);
#pragma unroll
    for (int j = 0; j < NT; ++j)
      acc[j] = __builtin_amdgcn_wmma_f32_16x16x32_bf16(false, a, false, bfr[j],
                                                       (short)0, acc[j], false, false);
    a = an;
#pragma unroll
    for (int j = 0; j < NT; ++j) bfr[j] = bn[j];
  }
#pragma unroll
  for (int j = 0; j < NT; ++j)
    acc[j] = __builtin_amdgcn_wmma_f32_16x16x32_bf16(false, a, false, bfr[j],
                                                     (short)0, acc[j], false, false);

  const int n = lane & 15;
  const int mbase = (lane >> 4) * 8;
#pragma unroll
  for (int j = 0; j < NT; ++j) {
    const int co = (ctBase + j) * 16 + n;
    const float bv = bias[co];
#pragma unroll
    for (int r = 0; r < 8; ++r) {
      int p2 = tloc * 16 + mbase + r;
      int oh2 = p2 >> woShift, ow2 = p2 & (Wo - 1);
      float v = acc[j][r] + bv;
      if (relu && v < 0.f) v = 0.f;
      out[(((((size_t)b * Ho + oh2) << woShift) + ow2) << coShift) + co] = f2bf(v);
    }
  }
}

// ---------------------------------------------------------------------------
// Implicit-GEMM transposed conv (k4 s2 p1) via parity decomposition, NHWC.
// grid.x = (B*4*tilesPerSub)/4, grid.y = (Co/16)/NT (or 1), block = 128
// ---------------------------------------------------------------------------
template <int NT>
__global__ __launch_bounds__(128)
void deconv2d_wmma(const bf16_t* __restrict__ in, const bf16_t* __restrict__ wpack,
                   const float* __restrict__ bias, bf16_t* __restrict__ outb,
                   float* __restrict__ outf,
                   int ciShift, int Hi, int Wi, int Ho, int wowShift,
                   int tpsShift, int nCoT, int coShift, int Co, int kTiles, int relu) {
  const int lane = threadIdx.x & 31;
  const int mt = blockIdx.x * 4 + (threadIdx.x >> 5);
  const int Ci = 1 << ciShift;
  const int WoW = 1 << wowShift;
  const int b = mt >> (tpsShift + 2);
  int rem = mt & ((4 << tpsShift) - 1);
  const int p = rem >> tpsShift;
  const int tloc = rem & ((1 << tpsShift) - 1);
  const int ph = p >> 1, pw = p & 1;
  const int ctBase = blockIdx.y * NT;
  const int m = lane & 15;
  const int pos = tloc * 16 + m;
  const int ohh = pos >> wowShift, oww = pos & (WoW - 1);
  const bf16_t* inB = in + (((size_t)b * Hi) << ciShift) * Wi;
  const bf16_t* wp = wpack + (size_t)p * kTiles * nCoT * 512;
  const int base8 = (lane >> 4) * 8;

  auto loadA = [&](int kt) -> v16bf {
    const int s0 = kt * 32 + base8;
    const int s1 = s0 + 16;
    int tap0 = s0 >> ciShift, ci0 = s0 & (Ci - 1);
    int ih0 = ohh + ph + (tap0 >> 1) - 1, iw0 = oww + pw + (tap0 & 1) - 1;
    bool ok0 = (unsigned)ih0 < (unsigned)Hi && (unsigned)iw0 < (unsigned)Wi;
    v4u r0 = load16_masked(inB, ((ih0 * Wi + iw0) << ciShift) + ci0, ok0);
    int tap1 = s1 >> ciShift, ci1 = s1 & (Ci - 1);
    int ih1 = ohh + ph + (tap1 >> 1) - 1, iw1 = oww + pw + (tap1 & 1) - 1;
    bool ok1 = (unsigned)ih1 < (unsigned)Hi && (unsigned)iw1 < (unsigned)Wi;
    v4u r1 = load16_masked(inB, ((ih1 * Wi + iw1) << ciShift) + ci1, ok1);
    return combine(r0, r1);
  };
  auto loadB = [&](int kt, int j) -> v16bf {
    return *(const v16bf*)(wp + ((size_t)(kt * nCoT + ctBase + j) * 32 + lane) * 16);
  };

  v8f acc[NT];
#pragma unroll
  for (int j = 0; j < NT; ++j)
#pragma unroll
    for (int i = 0; i < 8; ++i) acc[j][i] = 0.f;

  v16bf a = loadA(0);
  v16bf bfr[NT];
#pragma unroll
  for (int j = 0; j < NT; ++j) bfr[j] = loadB(0, j);

  for (int kt = 0; kt < kTiles - 1; ++kt) {
    v16bf an = loadA(kt + 1);
    v16bf bn[NT];
#pragma unroll
    for (int j = 0; j < NT; ++j) bn[j] = loadB(kt + 1, j);
#pragma unroll
    for (int j = 0; j < NT; ++j)
      acc[j] = __builtin_amdgcn_wmma_f32_16x16x32_bf16(false, a, false, bfr[j],
                                                       (short)0, acc[j], false, false);
    a = an;
#pragma unroll
    for (int j = 0; j < NT; ++j) bfr[j] = bn[j];
  }
#pragma unroll
  for (int j = 0; j < NT; ++j)
    acc[j] = __builtin_amdgcn_wmma_f32_16x16x32_bf16(false, a, false, bfr[j],
                                                     (short)0, acc[j], false, false);

  const int n = lane & 15;
  const int mbase = (lane >> 4) * 8;
  const int woShift = wowShift + 1;
  const int Wo = 1 << woShift;
#pragma unroll
  for (int j = 0; j < NT; ++j) {
    const int co = (ctBase + j) * 16 + n;
    if (co < Co) {
      const float bv = bias[co];
#pragma unroll
      for (int r = 0; r < 8; ++r) {
        int p2 = tloc * 16 + mbase + r;
        int oh = 2 * (p2 >> wowShift) + ph;
        int ow = 2 * (p2 & (WoW - 1)) + pw;
        float v = acc[j][r] + bv;
        if (relu && v < 0.f) v = 0.f;
        if (outf)  // final layer: NCHW f32
          outf[(((size_t)b * Co + co) * Ho + oh) * Wo + ow] = v;
        else       // NHWC bf16
          outb[(((((size_t)b * Ho + oh) << woShift) + ow) << coShift) + co] = f2bf(v);
      }
    }
  }
}

// ---------------------------------------------------------------------------
// Codebook scoring + argmax. feat NHWC [16*1024 tokens, 256ch].
// A fragments (K=256) preloaded once; 512 column tiles x 8 WMMA each,
// with B prefetched TWO linear steps ahead (partial loadcnt waits).
// grid.x = 1024/4, block = 128
// ---------------------------------------------------------------------------
__global__ __launch_bounds__(128)
void vq_argmax_wmma(const bf16_t* __restrict__ feat, const bf16_t* __restrict__ cbpack,
                    int* __restrict__ idxout, float* __restrict__ idxoutf) {
  const int lane = threadIdx.x & 31;
  const int mt = blockIdx.x * 4 + (threadIdx.x >> 5);
  const int m = lane & 15;
  const int token = mt * 16 + m;
  const bf16_t* fp = feat + (size_t)token * 256;
  const int base8 = (lane >> 4) * 8;

  v16bf a[8];
#pragma unroll
  for (int kt = 0; kt < 8; ++kt) {
    int s0 = kt * 32 + base8;
    v4u r0 = *(const v4u*)(fp + s0);
    v4u r1 = *(const v4u*)(fp + s0 + 16);
    a[kt] = combine(r0, r1);
  }

  // linear step s = ctile*8 + kt maps to B tile (kt*512 + ctile)
  auto loadB = [&](int s) -> v16bf {
    int ct = s >> 3, kt = s & 7;
    return *(const v16bf*)(cbpack + ((size_t)(kt * 512 + ct) * 32 + lane) * 16);
  };

  float bestv[8];
  int   besti[8];
#pragma unroll
  for (int r = 0; r < 8; ++r) { bestv[r] = -3.4e38f; besti[r] = 0; }

  // 2-deep prefetch: 4 b128 loads in flight -> partial waits before WMMA
  v16bf b0 = loadB(0);
  v16bf b1 = loadB(1);
  for (int ctile = 0; ctile < 512; ++ctile) {
    v8f acc;
#pragma unroll
    for (int i = 0; i < 8; ++i) acc[i] = 0.f;
#pragma unroll
    for (int kt = 0; kt < 8; ++kt) {
      int s = ctile * 8 + kt + 2;
      v16bf bn = loadB(s < 4096 ? s : (s - 4096));  // wrap to valid tile
      acc = __builtin_amdgcn_wmma_f32_16x16x32_bf16(false, a[kt], false, b0,
                                                    (short)0, acc, false, false);
      b0 = b1;
      b1 = bn;
    }
    int code = ctile * 16 + (lane & 15);
#pragma unroll
    for (int r = 0; r < 8; ++r) {
      float v = acc[r];
      if (v > bestv[r]) { bestv[r] = v; besti[r] = code; }
    }
  }

#pragma unroll
  for (int off = 8; off >= 1; off >>= 1) {
#pragma unroll
    for (int r = 0; r < 8; ++r) {
      float ov = __shfl_xor(bestv[r], off, 32);
      int   oi = __shfl_xor(besti[r], off, 32);
      if (ov > bestv[r] || (ov == bestv[r] && oi < besti[r])) {
        bestv[r] = ov; besti[r] = oi;
      }
    }
  }

  if ((lane & 15) == 0) {
    const int mbase = (lane >> 4) * 8;
#pragma unroll
    for (int r = 0; r < 8; ++r) {
      int tok2 = mt * 16 + mbase + r;
      idxout[tok2]  = besti[r];
      idxoutf[tok2] = (float)besti[r];
    }
  }
}

// ---------------------------------------------------------------------------
// Quantized gather: q[token, c] = cb[idx[token], c]  (NHWC bf16, coalesced)
// ---------------------------------------------------------------------------
__global__ void gather_q(const float* __restrict__ cb, const int* __restrict__ idx,
                         bf16_t* __restrict__ q, int n) {
  int i = blockIdx.x * blockDim.x + threadIdx.x;
  int stride = gridDim.x * blockDim.x;
  for (; i < n; i += stride) {
    int c = i & 255;
    int tok = i >> 8;
    q[i] = f2bf(cb[(size_t)idx[tok] * 256 + c]);
  }
}

// ===========================================================================
// Host-side orchestration
// ===========================================================================
extern "C" void kernel_launch(void* const* d_in, const int* in_sizes, int n_in,
                              void* d_out, int out_size, void* d_ws, size_t ws_size,
                              hipStream_t stream) {
  const float* x   = (const float*)d_in[0];
  const float* ew1 = (const float*)d_in[1];
  const float* eb1 = (const float*)d_in[2];
  const float* ew2 = (const float*)d_in[3];
  const float* eb2 = (const float*)d_in[4];
  const float* ew3 = (const float*)d_in[5];
  const float* eb3 = (const float*)d_in[6];
  const float* cb  = (const float*)d_in[7];
  const float* dw1 = (const float*)d_in[8];
  const float* db1 = (const float*)d_in[9];
  const float* dw2 = (const float*)d_in[10];
  const float* db2 = (const float*)d_in[11];
  const float* dw3 = (const float*)d_in[12];
  const float* db3 = (const float*)d_in[13];

  char* ws = (char*)d_ws;
  // --- workspace layout (bytes, 256-aligned; total ~71 MB) ---
  const size_t o_ew1 = 0;            //  16 KB  (K=128  x Co=64)
  const size_t o_ew2 = 16384;        // 256 KB  (K=1024 x Co=128)
  const size_t o_ew3 = 278528;       //   1 MB  (K=2048 x Co=256)
  const size_t o_dw1 = 1327104;      //   1 MB  (4 x K=1024 x 128)
  const size_t o_dw2 = 2375680;      // 256 KB  (4 x K=512 x 64)
  const size_t o_dw3 = 2637824;      //  32 KB  (4 x K=256 x 16pad)
  const size_t o_cb  = 2670592;      //   4 MB  (K=256 x N=8192)
  const size_t o_idx = 6864896;      //  64 KB
  const size_t o_A   = 7340032;      //  32 MB region A (xb/h2/q/d2)
  const size_t o_B   = 40894464;     //  32 MB region B (h1/feat/d1)

  bf16_t* p_ew1 = (bf16_t*)(ws + o_ew1);
  bf16_t* p_ew2 = (bf16_t*)(ws + o_ew2);
  bf16_t* p_ew3 = (bf16_t*)(ws + o_ew3);
  bf16_t* p_dw1 = (bf16_t*)(ws + o_dw1);
  bf16_t* p_dw2 = (bf16_t*)(ws + o_dw2);
  bf16_t* p_dw3 = (bf16_t*)(ws + o_dw3);
  bf16_t* p_cb  = (bf16_t*)(ws + o_cb);
  int*    idx   = (int*)(ws + o_idx);
  bf16_t* regA  = (bf16_t*)(ws + o_A);
  bf16_t* regB  = (bf16_t*)(ws + o_B);

  float* recon = (float*)d_out;                       // 16*3*256*256 NCHW
  float* idxf  = (float*)d_out + 16 * 3 * 256 * 256;  // 16*1024 indices

  // --- packs (B-fragment layout, channel-minor K) ---
  pack_conv_w  <<<dim3(4 * 4),      32, 0, stream>>>(ew1, p_ew1, 3, 3,  64,  128);
  pack_conv_w  <<<dim3(32 * 8),     32, 0, stream>>>(ew2, p_ew2, 6, 64, 128, 1024);
  pack_conv_w  <<<dim3(64 * 16),    32, 0, stream>>>(ew3, p_ew3, 7, 128, 256, 2048);
  pack_deconv_w<<<dim3(4 * 32 * 8), 32, 0, stream>>>(dw1, p_dw1, 8, 128, 1024);
  pack_deconv_w<<<dim3(4 * 16 * 4), 32, 0, stream>>>(dw2, p_dw2, 7, 64,  512);
  pack_deconv_w<<<dim3(4 * 8 * 1),  32, 0, stream>>>(dw3, p_dw3, 6, 3,   256);
  pack_cb      <<<dim3(8 * 512),    32, 0, stream>>>(cb, p_cb);

  // --- input -> NHWC bf16 padded to Ci=8 ---
  cvt_x_nhwc<<<dim3(8192), 256, 0, stream>>>(x, regA, 16 * 256 * 256 * 8);

  // --- encoder ---
  //                 in    wpack  bias out   ciS Hi  Wi   Ho  woS tpiS nCoT coS kT relu
  conv2d_wmma<4><<<dim3(4096, 1), 128, 0, stream>>>(
      regA, p_ew1, eb1, regB, 3, 256, 256, 128, 7, 10, 4, 6, 4, 1);
  conv2d_wmma<4><<<dim3(1024, 2), 128, 0, stream>>>(
      regB, p_ew2, eb2, regA, 6, 128, 128, 64, 6, 8, 8, 7, 32, 1);
  conv2d_wmma<4><<<dim3(256, 4), 128, 0, stream>>>(
      regA, p_ew3, eb3, regB, 7, 64, 64, 32, 5, 6, 16, 8, 64, 0);

  // --- codebook argmax (feat NHWC in regB) ---
  vq_argmax_wmma<<<dim3(256), 128, 0, stream>>>(regB, p_cb, idx, idxf);

  // --- quantized gather -> regA (NHWC) ---
  gather_q<<<dim3(8192), 256, 0, stream>>>(cb, idx, regA, 16 * 1024 * 256);

  // --- decoder ---
  //                  in    wpack  bias  outb  outf ciS Hi  Wi  Ho  wowS tpsS nCoT coS Co  kT relu
  deconv2d_wmma<4><<<dim3(1024, 2), 128, 0, stream>>>(
      regA, p_dw1, db1, regB, nullptr, 8, 32, 32, 64, 5, 6, 8, 7, 128, 32, 1);
  deconv2d_wmma<4><<<dim3(4096, 1), 128, 0, stream>>>(
      regB, p_dw2, db2, regA, nullptr, 7, 64, 64, 128, 6, 8, 4, 6, 64, 16, 1);
  deconv2d_wmma<1><<<dim3(16384, 1), 128, 0, stream>>>(
      regA, p_dw3, db3, nullptr, recon, 6, 128, 128, 256, 7, 10, 1, 0, 3, 8, 0);
}
